// GraphAutoencoder_74371653697942
// MI455X (gfx1250) — matrix operand order repaired and checked by reading, not compiled, not run on hardware
//
#include <hip/hip_runtime.h>

// GraphSAGE autoencoder on MI455X (gfx1250, wave32).
//
// Roofline: scatter-gather over 1.6M edges dominates (L2-resident: x=51MB,
// h=102MB < 192MB L2). GEMMs total ~30 GFLOP -> noise on WMMA. Layer-2
// aggregation is moved AFTER the lin_l projection (mean commutes with the
// linear map), shrinking that scatter from 1KB/edge to 256B/edge.

typedef _Float16 v16h __attribute__((ext_vector_type(16)));
typedef float    v8f  __attribute__((ext_vector_type(8)));

#define N_NODES 100000

// ---- WMMA fragment loaders (ISA 7.12.2 16-bit layouts), f32 -> f16 convert ----

// A 16x32 (MxK): lane g = lane>>4; VGPR v<4 holds K = g*8 + v*2 (+1),
// v>=4 holds K = 16 + g*8 + (v-4)*2 (+1). Row = lane&15.
__device__ __forceinline__ v16h load_a_frag(const float* __restrict__ A, int ldk,
                                            int row, int k0, int g, float scale) {
    const float* p = A + (size_t)row * ldk + k0 + g * 8;
    v16h a;
#pragma unroll
    for (int j = 0; j < 16; ++j) {
        int vg = j >> 1, w = j & 1;
        int kk = (vg < 4 ? vg * 2 : 16 + (vg - 4) * 2) + w;
        a[j] = (_Float16)(p[kk] * scale);
    }
    return a;
}

// B 32x16 (KxN): lane n = lane&15, g = lane>>4;
// element j<8 maps to K = g*8 + j, j>=8 to K = 16 + g*8 + (j-8) (matches A).
__device__ __forceinline__ v16h load_b_frag(const float* __restrict__ W, int ldk,
                                            int col, int k0, int g) {
    const float* p = W + (size_t)col * ldk + k0 + g * 8;
    v16h b;
#pragma unroll
    for (int j = 0; j < 16; ++j) b[j] = (_Float16)p[j < 8 ? j : j + 8];
    return b;
}

// ---- Unified GEMM: out = epilogue( A1*W1^T [+ A2*W2^T] [+ Eadd/deg] [+ bias] )
// A1 rows optionally pre-scaled by 1/max(deg,1) (mean aggregation).
// One wave computes a 32x16 tile (2 accumulators, B frags reused).
__global__ __launch_bounds__(256) void gemm_wmma_kernel(
    const float* __restrict__ A1, const float* __restrict__ W1,
    const float* __restrict__ A2, const float* __restrict__ W2,
    const float* __restrict__ Eadd, const float* __restrict__ bias,
    const float* __restrict__ deg, float* __restrict__ out,
    int M, int N, int K, int scaleA1, int relu)
{
    int wave = (int)((blockIdx.x * blockDim.x + threadIdx.x) >> 5);
    int lane = threadIdx.x & 31;
    int tiles_n = N >> 4;
    int total   = (M >> 5) * tiles_n;          // M divisible by 32, N by 16
    if (wave >= total) return;                 // wave-uniform: EXEC all-1s inside

    int tm = wave / tiles_n, tn = wave % tiles_n;
    int m0 = tm << 5, n0 = tn << 4;
    int g = lane >> 4, r = lane & 15;
    int rowA = m0 + r, rowB = m0 + 16 + r;

    float s0 = 1.f, s1 = 1.f;
    if (scaleA1) {
        s0 = 1.f / fmaxf(deg[rowA], 1.f);
        s1 = 1.f / fmaxf(deg[rowB], 1.f);
    }

    v8f acc0 = {}; v8f acc1 = {};
    for (int k0 = 0; k0 < K; k0 += 32) {
        v16h b1 = load_b_frag(W1, K, n0 + r, k0, g);
        v16h a0 = load_a_frag(A1, K, rowA, k0, g, s0);
        v16h a1 = load_a_frag(A1, K, rowB, k0, g, s1);
        acc0 = __builtin_amdgcn_wmma_f32_16x16x32_f16(false, a0, false, b1,
                                                      (short)0, acc0, false, false);
        acc1 = __builtin_amdgcn_wmma_f32_16x16x32_f16(false, a1, false, b1,
                                                      (short)0, acc1, false, false);
        if (A2) {  // fused second GEMM (layer 1: x @ W_r^T)
            v16h b2 = load_b_frag(W2, K, n0 + r, k0, g);
            v16h c0 = load_a_frag(A2, K, rowA, k0, g, 1.f);
            v16h c1 = load_a_frag(A2, K, rowB, k0, g, 1.f);
            acc0 = __builtin_amdgcn_wmma_f32_16x16x32_f16(false, c0, false, b2,
                                                          (short)0, acc0, false, false);
            acc1 = __builtin_amdgcn_wmma_f32_16x16x32_f16(false, c1, false, b2,
                                                          (short)0, acc1, false, false);
        }
    }

    // C/D layout: acc[i] -> (m = m0 + i + 8*g, n = n0 + (lane&15))
    int n = n0 + r;
    float bv = bias ? bias[n] : 0.f;
#pragma unroll
    for (int i = 0; i < 8; ++i) {
        int m = m0 + i + g * 8;
        float v = acc0[i] + bv;
        if (Eadd) v += Eadd[(size_t)m * N + n] / fmaxf(deg[m], 1.f);
        if (relu) v = fmaxf(v, 0.f);
        out[(size_t)m * N + n] = v;
    }
#pragma unroll
    for (int i = 0; i < 8; ++i) {
        int m = m0 + 16 + i + g * 8;
        float v = acc1[i] + bv;
        if (Eadd) v += Eadd[(size_t)m * N + n] / fmaxf(deg[m], 1.f);
        if (relu) v = fmaxf(v, 0.f);
        out[(size_t)m * N + n] = v;
    }
}

// ---- Layer-1 scatter: one wave per edge, 128 floats (float4/lane) + degree ----
__global__ __launch_bounds__(256) void scatter_x_deg_kernel(
    const float* __restrict__ x, const int* __restrict__ ei, int E,
    float* __restrict__ agg, float* __restrict__ deg)
{
    int e = (int)((blockIdx.x * blockDim.x + threadIdx.x) >> 5);
    int lane = threadIdx.x & 31;
    if (e >= E) return;
    int src = ei[e];
    int dst = ei[E + e];
    float4 v = ((const float4*)(x + (size_t)src * 128))[lane];
    float* d = agg + (size_t)dst * 128 + lane * 4;
    atomicAdd(d + 0, v.x); atomicAdd(d + 1, v.y);
    atomicAdd(d + 2, v.z); atomicAdd(d + 3, v.w);
    if (lane == 0) atomicAdd(deg + dst, 1.0f);
}

// ---- Layer-2 scatter of projected 64-d vectors (float2/lane) ----
__global__ __launch_bounds__(256) void scatter_p2_kernel(
    const float* __restrict__ p2, const int* __restrict__ ei, int E,
    float* __restrict__ agg)
{
    int e = (int)((blockIdx.x * blockDim.x + threadIdx.x) >> 5);
    int lane = threadIdx.x & 31;
    if (e >= E) return;
    int src = ei[e];
    int dst = ei[E + e];
    float2 v = ((const float2*)(p2 + (size_t)src * 64))[lane];
    float* d = agg + (size_t)dst * 64 + lane * 2;
    atomicAdd(d + 0, v.x); atomicAdd(d + 1, v.y);
}

extern "C" void kernel_launch(void* const* d_in, const int* in_sizes, int n_in,
                              void* d_out, int out_size, void* d_ws, size_t ws_size,
                              hipStream_t stream) {
    const float* x    = (const float*)d_in[0];
    const int*   ei   = (const int*)d_in[1];   // [2,E] (int32 on device)
    const float* W_l1 = (const float*)d_in[2];
    const float* b_l1 = (const float*)d_in[3];
    const float* W_r1 = (const float*)d_in[4];
    const float* W_l2 = (const float*)d_in[5];
    const float* b_l2 = (const float*)d_in[6];
    const float* W_r2 = (const float*)d_in[7];
    const float* fc1W = (const float*)d_in[8];
    const float* fc1b = (const float*)d_in[9];
    const float* fc2W = (const float*)d_in[10];
    const float* fc2b = (const float*)d_in[11];

    const int E = in_sizes[1] / 2;

    // Workspace layout (floats):
    //   deg   [100k]            @ 0
    //   regionA [25.6M]         @ 102400 : agg1(12.8M) | p2(6.4M) | agg2(6.4M)
    //                                      -> reused wholly as hid(25.6M)
    //   h     [25.6M]           after regionA
    float* ws      = (float*)d_ws;
    float* deg     = ws;
    float* regionA = ws + 102400;
    float* agg1    = regionA;
    float* p2      = regionA + 12800000;
    float* agg2    = regionA + 19200000;
    float* hid     = regionA;                     // reuse after emb is done
    float* h       = regionA + 25600000;
    float* emb     = (float*)d_out;               // [100k, 64]
    float* recon   = (float*)d_out + (size_t)N_NODES * 64;  // [100k, 128]

    // Deterministic per-call init of accumulators (graph-capturable memsets).
    hipMemsetAsync(deg,  0, (size_t)N_NODES * sizeof(float), stream);
    hipMemsetAsync(agg1, 0, (size_t)N_NODES * 128 * sizeof(float), stream);
    hipMemsetAsync(agg2, 0, (size_t)N_NODES * 64 * sizeof(float), stream);

    const int sblocks = (E + 7) / 8;  // 8 waves/block, 1 edge/wave

    auto gemm = [&](const float* A1, const float* W1, const float* A2,
                    const float* W2, const float* Eadd, const float* bias,
                    float* out, int N, int K, int scaleA1, int relu) {
        int tiles  = (N_NODES >> 5) * (N >> 4);
        int blocks = (tiles + 7) / 8;
        gemm_wmma_kernel<<<blocks, 256, 0, stream>>>(A1, W1, A2, W2, Eadd, bias,
                                                     deg, out, N_NODES, N, K,
                                                     scaleA1, relu);
    };

    // Layer 1: agg1 = sum_edges x[src]; deg
    scatter_x_deg_kernel<<<sblocks, 256, 0, stream>>>(x, ei, E, agg1, deg);
    // h = relu((agg1/deg) @ W_l1^T + x @ W_r1^T + b_l1)   [100k,256]
    gemm(agg1, W_l1, x, W_r1, nullptr, b_l1, h, 256, 128, /*scaleA1=*/1, /*relu=*/1);

    // Layer 2 (projection-first): p2 = h @ W_l2^T          [100k,64]
    gemm(h, W_l2, nullptr, nullptr, nullptr, nullptr, p2, 64, 256, 0, 0);
    // agg2 = sum_edges p2[src]
    scatter_p2_kernel<<<sblocks, 256, 0, stream>>>(p2, ei, E, agg2);
    // emb = h @ W_r2^T + agg2/deg + b_l2                   [100k,64] -> d_out
    gemm(h, W_r2, nullptr, nullptr, agg2, b_l2, emb, 64, 256, 0, 0);

    // Decoder
    // hid = relu(emb @ fc1_W^T + fc1_b)                    [100k,256]
    gemm(emb, fc1W, nullptr, nullptr, nullptr, fc1b, hid, 256, 64, 0, 1);
    // recon = hid @ fc2_W^T + fc2_b                        [100k,128] -> d_out
    gemm(hid, fc2W, nullptr, nullptr, nullptr, fc2b, recon, 128, 256, 0, 0);
}
